// BasicTransformerBlock_50208167690869
// MI455X (gfx1250) — matrix operand order repaired and checked by reading, not compiled, not run on hardware
//
#include <hip/hip_runtime.h>

// ---------------------------------------------------------------------------
// Problem constants (reference shapes: B=16 frames, S=1024, D=640, 8 heads x 80)
// ---------------------------------------------------------------------------
#define BATCH   16
#define SEQ     1024
#define DMODEL  640
#define HEADS   8
#define DH      80
#define KTOP    512            // max(64, S//2)
#define ATT_SCALE 0.11180339887498948f   // 80^-0.5

typedef _Float16 h16;
typedef __attribute__((ext_vector_type(16))) _Float16 v16h;
typedef __attribute__((ext_vector_type(8)))  _Float16 h8v;
typedef __attribute__((ext_vector_type(8)))  float    v8f;

union Frag {
    v16h v;
    _Float16 e[16];
    h8v h8[2];
};

__device__ inline v8f vzero8f() {
    v8f z;
    #pragma unroll
    for (int i = 0; i < 8; ++i) z[i] = 0.0f;
    return z;
}

__device__ inline h8v hzero8() {
    h8v z;
    #pragma unroll
    for (int i = 0; i < 8; ++i) z[i] = (_Float16)0.0f;
    return z;
}

__device__ inline v8f wmma16x16x32(const Frag& a, const Frag& b, v8f c) {
    // D = A(16x32 f16) * B(32x16 f16) + C(16x16 f32)
    return __builtin_amdgcn_wmma_f32_16x16x32_f16(
        false, a.v, false, b.v, (short)0, c, false, false);
}

// ---------------------------------------------------------------------------
// 1) prep: hidden f32 -> f16, and diff_score[b,s] = mean(|hidden - pre|)
// ---------------------------------------------------------------------------
__global__ __launch_bounds__(128)
void prep_kernel(const float* __restrict__ hs, h16* __restrict__ hsb,
                 float* __restrict__ scores) {
    int row = blockIdx.x;                 // 0 .. BATCH*SEQ-1
    int b = row / SEQ;
    int s = row - b * SEQ;
    int prow = (b == 0 ? 0 : (b - 1)) * SEQ + s;
    const float* cur = hs + (size_t)row * DMODEL;
    const float* pre = hs + (size_t)prow * DMODEL;
    float acc = 0.0f;
    for (int d = threadIdx.x; d < DMODEL; d += blockDim.x) {
        float c = cur[d];
        hsb[(size_t)row * DMODEL + d] = (h16)c;
        acc += fabsf(c - pre[d]);
    }
    __shared__ float red[4];
    #pragma unroll
    for (int off = 16; off; off >>= 1) acc += __shfl_xor(acc, off, 32);
    if ((threadIdx.x & 31) == 0) red[threadIdx.x >> 5] = acc;
    __syncthreads();
    if (threadIdx.x == 0)
        scores[row] = (red[0] + red[1] + red[2] + red[3]) * (1.0f / DMODEL);
}

// ---------------------------------------------------------------------------
// 2) top-K: bitonic sort of 1024 (score, idx), descending, index tie-break
// ---------------------------------------------------------------------------
__global__ __launch_bounds__(512)
void topk_kernel(const float* __restrict__ scores, int* __restrict__ topk) {
    __shared__ float s[SEQ];
    __shared__ int   id[SEQ];
    int b = blockIdx.x;
    for (int i = threadIdx.x; i < SEQ; i += blockDim.x) {
        s[i] = scores[b * SEQ + i];
        id[i] = i;
    }
    __syncthreads();
    for (int size = 2; size <= SEQ; size <<= 1) {
        for (int stride = size >> 1; stride > 0; stride >>= 1) {
            for (int t = threadIdx.x; t < SEQ / 2; t += blockDim.x) {
                int lo = 2 * t - (t & (stride - 1));
                int hi = lo + stride;
                bool desc = ((lo & size) == 0);
                float slo = s[lo], shi = s[hi];
                int   ilo = id[lo], ihi = id[hi];
                bool lo_first = (slo > shi) || (slo == shi && ilo < ihi);
                bool sw = desc ? !lo_first : lo_first;
                if (sw) { s[lo] = shi; s[hi] = slo; id[lo] = ihi; id[hi] = ilo; }
            }
            __syncthreads();
        }
    }
    for (int j = threadIdx.x; j < KTOP; j += blockDim.x)
        topk[b * KTOP + j] = id[j];
}

// ---------------------------------------------------------------------------
// 3) small elementwise helpers
// ---------------------------------------------------------------------------
__global__ void cvt_kernel(const float* __restrict__ src, h16* __restrict__ dst, int n) {
    int i = blockIdx.x * blockDim.x + threadIdx.x;
    if (i < n) dst[i] = (h16)src[i];
}

__global__ __launch_bounds__(128)
void gather_diff_kernel(const float* __restrict__ hs, const int* __restrict__ topk,
                        h16* __restrict__ gd) {
    int r = blockIdx.x;                   // 0 .. BATCH*KTOP-1
    int b = r / KTOP;
    int idx = topk[r];
    int prev = (b == 0) ? 0 : b - 1;
    const float* cur = hs + ((size_t)b * SEQ + idx) * DMODEL;
    const float* pre = hs + ((size_t)prev * SEQ + idx) * DMODEL;
    for (int d = threadIdx.x; d < DMODEL; d += blockDim.x)
        gd[(size_t)r * DMODEL + d] = (h16)(cur[d] - pre[d]);
}

__global__ __launch_bounds__(128)
void phs_kernel(const float* __restrict__ p1, const float* __restrict__ bias,
                const float* __restrict__ hs, const int* __restrict__ topk,
                h16* __restrict__ phs) {
    int r = blockIdx.x;
    int b = r / KTOP;
    int idx = topk[r];
    const float* g = hs + ((size_t)b * SEQ + idx) * DMODEL;
    for (int d = threadIdx.x; d < DMODEL; d += blockDim.x)
        phs[(size_t)r * DMODEL + d] = (h16)(p1[(size_t)r * DMODEL + d] + bias[d] + g[d]);
}

__global__ void cumsum_kernel(float* __restrict__ comp) {
    int pos = blockIdx.x * blockDim.x + threadIdx.x;  // KTOP*DMODEL positions
    if (pos >= KTOP * DMODEL) return;
    float acc = 0.0f;
    for (int f = 0; f < BATCH; ++f) {
        size_t off = (size_t)f * KTOP * DMODEL + pos;
        acc += comp[off];
        comp[off] = acc;
    }
}

__global__ __launch_bounds__(128)
void scatter_add_kernel(const float* __restrict__ comp, const int* __restrict__ topk,
                        float* __restrict__ out) {
    int r = blockIdx.x;
    int b = r / KTOP;
    int row = topk[r];
    float* o = out + ((size_t)b * SEQ + row) * DMODEL;
    const float* c = comp + (size_t)r * DMODEL;
    for (int d = threadIdx.x; d < DMODEL; d += blockDim.x)
        o[d] += c[d];
}

// ---------------------------------------------------------------------------
// 4) Tiled WMMA GEMM:  C[M,N] = A[M,K] @ W[K,N] (+bias)
//    Block tile 64x64, 4 waves, K-panel 32; W staged transposed in LDS so
//    all fragments are two ds_load_b128.  M%64==N%64==K%32==0 everywhere.
// ---------------------------------------------------------------------------
template <bool OUT_F32>
__global__ __launch_bounds__(128)
void gemm_kernel(const h16* __restrict__ A, const h16* __restrict__ W,
                 const float* __restrict__ bias, void* __restrict__ Cv,
                 int M, int N, int K) {
    __shared__ h16 As[64][32];   // A tile, row-major
    __shared__ h16 Wt[64][32];   // W tile, transposed: Wt[n][k]

    const int n0 = blockIdx.x * 64;
    const int m0 = blockIdx.y * 64;
    const int tid = threadIdx.x;
    const int lane = tid & 31, w = tid >> 5;
    const int hl = lane >> 4, ln = lane & 15;

    v8f acc[4];
    #pragma unroll
    for (int t = 0; t < 4; ++t) acc[t] = vzero8f();

    for (int k0 = 0; k0 < K; k0 += 32) {
        __syncthreads();
        {   // A: 128 threads x 16 elems (two b128 global loads each)
            int r = tid >> 1, c = (tid & 1) * 16;
            const h8v* src = (const h8v*)(A + (size_t)(m0 + r) * K + k0 + c);
            *(h8v*)&As[r][c]     = src[0];
            *(h8v*)&As[r][c + 8] = src[1];
            if (k0 + 32 < K)
                __builtin_prefetch(A + (size_t)(m0 + r) * K + k0 + 32 + c, 0, 1);
        }
        {   // W: read row-major, write transposed
            int r = tid >> 2, c = (tid & 3) * 16;
            const h16* src = W + (size_t)(k0 + r) * N + n0 + c;
            #pragma unroll
            for (int i = 0; i < 16; ++i) Wt[c + i][r] = src[i];
            if (k0 + 32 < K)
                __builtin_prefetch(W + (size_t)(k0 + 32 + r) * N + n0 + c, 0, 1);
        }
        __syncthreads();

        Frag af;
        af.h8[0] = *(const h8v*)&As[w * 16 + ln][hl * 16];
        af.h8[1] = *(const h8v*)&As[w * 16 + ln][hl * 16 + 8];
        #pragma unroll
        for (int t = 0; t < 4; ++t) {
            Frag bf;
            bf.h8[0] = *(const h8v*)&Wt[t * 16 + ln][hl * 16];
            bf.h8[1] = *(const h8v*)&Wt[t * 16 + ln][hl * 16 + 8];
            acc[t] = wmma16x16x32(af, bf, acc[t]);
        }
    }

    // C layout: element r -> M = r + 8*(lane>=16), N = lane&15
    const int m_base = m0 + w * 16 + hl * 8;
    #pragma unroll
    for (int t = 0; t < 4; ++t) {
        int col = n0 + t * 16 + ln;
        float bv = bias ? bias[col] : 0.0f;
        #pragma unroll
        for (int r = 0; r < 8; ++r) {
            float val = acc[t][r] + bv;
            size_t off = (size_t)(m_base + r) * N + col;
            if (OUT_F32) ((float*)Cv)[off] = val;
            else         ((h16*)Cv)[off] = (h16)val;
        }
    }
}

// ---------------------------------------------------------------------------
// 4b) GEMM with transposed f16 output:  VT[b][n][s] = (A @ W + bias)^T
//     Used for the V projections so attention's PV B-operand is contiguous
//     in K.  C tile is bounced through padded LDS (row pitch 144B, 16B
//     aligned) so the global stores are fully vectorized b128.
//     Lb = rows per batch in A (1024 here); 64-row tiles never straddle.
// ---------------------------------------------------------------------------
__global__ __launch_bounds__(128)
void gemm_tn_kernel(const h16* __restrict__ A, const h16* __restrict__ W,
                    const float* __restrict__ bias, h16* __restrict__ VT,
                    int M, int N, int K, int Lb) {
    __shared__ h16 As[64][32];
    __shared__ h16 Wt[64][32];
    __shared__ h16 ct[64][72];   // C^T tile: ct[n][m], pitch 72 h16 = 144B

    const int n0 = blockIdx.x * 64;
    const int m0 = blockIdx.y * 64;
    const int tid = threadIdx.x;
    const int lane = tid & 31, w = tid >> 5;
    const int hl = lane >> 4, ln = lane & 15;

    v8f acc[4];
    #pragma unroll
    for (int t = 0; t < 4; ++t) acc[t] = vzero8f();

    for (int k0 = 0; k0 < K; k0 += 32) {
        __syncthreads();
        {
            int r = tid >> 1, c = (tid & 1) * 16;
            const h8v* src = (const h8v*)(A + (size_t)(m0 + r) * K + k0 + c);
            *(h8v*)&As[r][c]     = src[0];
            *(h8v*)&As[r][c + 8] = src[1];
        }
        {
            int r = tid >> 2, c = (tid & 3) * 16;
            const h16* src = W + (size_t)(k0 + r) * N + n0 + c;
            #pragma unroll
            for (int i = 0; i < 16; ++i) Wt[c + i][r] = src[i];
        }
        __syncthreads();

        Frag af;
        af.h8[0] = *(const h8v*)&As[w * 16 + ln][hl * 16];
        af.h8[1] = *(const h8v*)&As[w * 16 + ln][hl * 16 + 8];
        #pragma unroll
        for (int t = 0; t < 4; ++t) {
            Frag bf;
            bf.h8[0] = *(const h8v*)&Wt[t * 16 + ln][hl * 16];
            bf.h8[1] = *(const h8v*)&Wt[t * 16 + ln][hl * 16 + 8];
            acc[t] = wmma16x16x32(af, bf, acc[t]);
        }
    }

    // stage C transposed into LDS
    __syncthreads();
    const int m_loc = w * 16 + hl * 8;           // local m of acc element r=0
    #pragma unroll
    for (int t = 0; t < 4; ++t) {
        int nloc = t * 16 + ln;
        float bv = bias ? bias[n0 + nloc] : 0.0f;
        #pragma unroll
        for (int r = 0; r < 8; ++r)
            ct[nloc][m_loc + r] = (h16)(acc[t][r] + bv);
    }
    __syncthreads();

    // vectorized transposed store: VT[(b*DMODEL + n)*Lb + m]
    const int bb = m0 / Lb;
    const int ml = m0 - bb * Lb;
    {
        int nloc = tid >> 1, mseg = (tid & 1) * 32;
        h16* dst = VT + ((size_t)bb * DMODEL + n0 + nloc) * Lb + ml + mseg;
        #pragma unroll
        for (int i = 0; i < 4; ++i)
            *(h8v*)(dst + i * 8) = *(const h8v*)&ct[nloc][mseg + i * 8];
    }
}

// ---------------------------------------------------------------------------
// 5) Flash attention.  grid = (Lq/16, HEADS, BATCH), block = 160 (5 waves).
//    K/V from batch max(b-1,0) ("former frame" / "pre" semantics).
//    V comes pre-transposed (VT[b][n][s]) so the PV loop is barrier-free:
//    A = two ds_load_b128 from packed f16 probs, B = two global_load_b128.
// ---------------------------------------------------------------------------
__global__ __launch_bounds__(160)
void attn_kernel(const h16* __restrict__ Q, const h16* __restrict__ Kp,
                 const h16* __restrict__ VT, h16* __restrict__ O, int Lq) {
    const int qt = blockIdx.x, h = blockIdx.y, b = blockIdx.z;
    const int kvb = (b == 0) ? 0 : b - 1;
    const int tid = threadIdx.x;
    const int lane = tid & 31, w = tid >> 5;
    const int hl = lane >> 4, ln = lane & 15;

    __shared__ h16   qs[16][96];      // q tile, head dim zero-padded 80 -> 96
    __shared__ float sc[16][256];     // raw scores (f32)
    __shared__ h16   scp[16][256];    // probs packed f16 (PV A operand)
    __shared__ float st_m[16], st_l[16], st_f[16];

    const h16* qbase = Q + ((size_t)b * Lq + (size_t)qt * 16) * DMODEL + h * DH;
    for (int i = tid; i < 16 * 96; i += 160) {
        int r = i / 96, c = i % 96;
        qs[r][c] = (c < DH) ? qbase[(size_t)r * DMODEL + c] : (h16)0.0f;
    }
    if (tid < 16) { st_m[tid] = -1e30f; st_l[tid] = 0.0f; }
    v8f oacc = vzero8f();
    __syncthreads();

    const h16* kb  = Kp + (size_t)kvb * SEQ * DMODEL + h * DH;
    const h16* vtb = VT + ((size_t)kvb * DMODEL + h * DH) * SEQ;  // VT rows: head cols

    for (int c0 = 0; c0 < SEQ; c0 += 256) {
        // ---- scores S = q @ k^T * scale ------------------------------------
        for (int nt = w; nt < 16; nt += 5) {
            v8f acc = vzero8f();
            #pragma unroll
            for (int ks = 0; ks < 3; ++ks) {          // K = 96 (padded)
                Frag af, bf;
                af.h8[0] = *(const h8v*)&qs[ln][ks * 32 + hl * 16];
                af.h8[1] = *(const h8v*)&qs[ln][ks * 32 + hl * 16 + 8];
                int kcol = ks * 32 + hl * 16;         // B: K = kcol + e, N = ln
                if (kcol < DH) {
                    const h16* kr = kb + (size_t)(c0 + nt * 16 + ln) * DMODEL + kcol;
                    bf.h8[0] = *(const h8v*)kr;
                    bf.h8[1] = *(const h8v*)(kr + 8);
                } else {
                    bf.h8[0] = hzero8();
                    bf.h8[1] = hzero8();
                }
                acc = wmma16x16x32(af, bf, acc);
            }
            #pragma unroll
            for (int r = 0; r < 8; ++r)
                sc[r + hl * 8][nt * 16 + ln] = acc[r] * ATT_SCALE;
        }
        __syncthreads();

        // ---- streaming softmax update (emit packed f16 probs) ---------------
        for (int row = w; row < 16; row += 5) {
            float mx = -1e30f;
            for (int j = lane; j < 256; j += 32) mx = fmaxf(mx, sc[row][j]);
            #pragma unroll
            for (int off = 16; off; off >>= 1) mx = fmaxf(mx, __shfl_xor(mx, off, 32));
            float mold = st_m[row];
            float mnew = fmaxf(mold, mx);
            float sum = 0.0f;
            for (int j = lane; j < 256; j += 32) {
                float p = __expf(sc[row][j] - mnew);
                scp[row][j] = (h16)p;
                sum += p;
            }
            #pragma unroll
            for (int off = 16; off; off >>= 1) sum += __shfl_xor(sum, off, 32);
            if (lane == 0) {
                float f = __expf(mold - mnew);
                st_f[row] = f;
                st_l[row] = st_l[row] * f + sum;
                st_m[row] = mnew;
            }
        }
        __syncthreads();

        // rescale running output by exp(m_old - m_new)
        #pragma unroll
        for (int r = 0; r < 8; ++r) oacc[r] *= st_f[r + hl * 8];

        // ---- O += P @ V (barrier-free; V^T is contiguous in K) --------------
        for (int k0 = 0; k0 < 256; k0 += 32) {
            Frag af, bf;
            af.h8[0] = *(const h8v*)&scp[ln][k0 + hl * 16];
            af.h8[1] = *(const h8v*)&scp[ln][k0 + hl * 16 + 8];
            const h16* vr = vtb + (size_t)(w * 16 + ln) * SEQ + c0 + k0 + hl * 16;
            bf.h8[0] = *(const h8v*)vr;
            bf.h8[1] = *(const h8v*)(vr + 8);
            oacc = wmma16x16x32(af, bf, oacc);
        }
        __syncthreads();   // protect sc/scp reuse by next chunk
    }

    // normalize and store merged output (head h occupies cols h*80..h*80+79)
    h16* ob = O + ((size_t)b * Lq + (size_t)qt * 16) * DMODEL + h * DH + w * 16;
    #pragma unroll
    for (int r = 0; r < 8; ++r) {
        int m = r + hl * 8;
        ob[(size_t)m * DMODEL + ln] = (h16)(oacc[r] / st_l[m]);
    }
}

// ---------------------------------------------------------------------------
// Host orchestration
// ---------------------------------------------------------------------------
extern "C" void kernel_launch(void* const* d_in, const int* in_sizes, int n_in,
                              void* d_out, int out_size, void* d_ws, size_t ws_size,
                              hipStream_t stream) {
    (void)in_sizes; (void)n_in; (void)out_size; (void)ws_size;

    const float* hidden    = (const float*)d_in[0];
    const float* w_primary = (const float*)d_in[1];
    const float* b_primary = (const float*)d_in[2];
    const float* w_cq = (const float*)d_in[3];  const float* b_cq = (const float*)d_in[4];
    const float* w_ck = (const float*)d_in[5];  const float* b_ck = (const float*)d_in[6];
    const float* w_cv = (const float*)d_in[7];  const float* b_cv = (const float*)d_in[8];
    const float* w_co = (const float*)d_in[9];  const float* b_co = (const float*)d_in[10];
    const float* w_tq = (const float*)d_in[11];
    const float* w_tk = (const float*)d_in[12];
    const float* w_tv = (const float*)d_in[13];
    const float* w_to = (const float*)d_in[14]; const float* b_to = (const float*)d_in[15];
    float* out = (float*)d_out;

    // ---- workspace layout (aliased; main branch runs first) ----------------
    const size_t WB = (size_t)DMODEL * DMODEL;           // weight elements
    const size_t ROWS = (size_t)BATCH * SEQ;             // 16384
    const size_t KROWS = (size_t)BATCH * KTOP;           // 8192
    char* base = (char*)d_ws;
    size_t off = 0;
    auto salloc = [&](size_t bytes) { size_t o = off; off += (bytes + 255) & ~(size_t)255; return o; };

    h16*  wts    = (h16*) (base + salloc(9 * WB * sizeof(h16)));
    h16*  hsb    = (h16*) (base + salloc(ROWS * DMODEL * sizeof(h16)));
    float* scrs  = (float*)(base + salloc(ROWS * sizeof(float)));
    int*  topk   = (int*)  (base + salloc(KROWS * sizeof(int)));
    h16*  q2     = (h16*) (base + salloc(ROWS * DMODEL * sizeof(h16)));   // later: gdiff | phs
    h16*  k2     = (h16*) (base + salloc(ROWS * DMODEL * sizeof(h16)));   // later: ck
    h16*  v2t    = (h16*) (base + salloc(ROWS * DMODEL * sizeof(h16)));   // V^T; later: cv^T
    h16*  mattn  = (h16*) (base + salloc(ROWS * DMODEL * sizeof(h16)));   // later: cq | cattn
    float* comp  = (float*)(base + salloc(KROWS * DMODEL * sizeof(float)));

    h16* w_primary_h = wts + 0 * WB;
    h16* w_cq_h = wts + 1 * WB;  h16* w_ck_h = wts + 2 * WB;  h16* w_cv_h = wts + 3 * WB;
    h16* w_co_h = wts + 4 * WB;  h16* w_tq_h = wts + 5 * WB;  h16* w_tk_h = wts + 6 * WB;
    h16* w_tv_h = wts + 7 * WB;  h16* w_to_h = wts + 8 * WB;

    h16* gdiff = q2;                            // [8192,640] f16
    h16* phs   = q2 + KROWS * DMODEL;           // [8192,640] f16
    h16* cq    = mattn;                         // [8192,640] f16
    h16* cattn = mattn + KROWS * DMODEL;        // [8192,640] f16
    h16* ck    = k2;
    h16* cvt_t = v2t;                           // comp-V^T reuses the V^T region

    const int cvt_blk = 256, cvt_grid = (int)((WB + cvt_blk - 1) / cvt_blk);
    dim3 gemm_grid_full(DMODEL / 64, (int)(ROWS / 64));   // M=16384
    dim3 gemm_grid_half(DMODEL / 64, (int)(KROWS / 64));  // M=8192

    // ---- weight conversions -------------------------------------------------
    cvt_kernel<<<cvt_grid, cvt_blk, 0, stream>>>(w_primary, w_primary_h, (int)WB);
    cvt_kernel<<<cvt_grid, cvt_blk, 0, stream>>>(w_cq, w_cq_h, (int)WB);
    cvt_kernel<<<cvt_grid, cvt_blk, 0, stream>>>(w_ck, w_ck_h, (int)WB);
    cvt_kernel<<<cvt_grid, cvt_blk, 0, stream>>>(w_cv, w_cv_h, (int)WB);
    cvt_kernel<<<cvt_grid, cvt_blk, 0, stream>>>(w_co, w_co_h, (int)WB);
    cvt_kernel<<<cvt_grid, cvt_blk, 0, stream>>>(w_tq, w_tq_h, (int)WB);
    cvt_kernel<<<cvt_grid, cvt_blk, 0, stream>>>(w_tk, w_tk_h, (int)WB);
    cvt_kernel<<<cvt_grid, cvt_blk, 0, stream>>>(w_tv, w_tv_h, (int)WB);
    cvt_kernel<<<cvt_grid, cvt_blk, 0, stream>>>(w_to, w_to_h, (int)WB);

    // ---- prep + top-K -------------------------------------------------------
    prep_kernel<<<(int)ROWS, 128, 0, stream>>>(hidden, hsb, scrs);
    topk_kernel<<<BATCH, 512, 0, stream>>>(scrs, topk);

    // ---- main branch: q2/k2/v2^T projections, attention, out projection ----
    gemm_kernel<false><<<gemm_grid_full, 128, 0, stream>>>(hsb, w_tq_h, nullptr, q2,
                                                           (int)ROWS, DMODEL, DMODEL);
    gemm_kernel<false><<<gemm_grid_full, 128, 0, stream>>>(hsb, w_tk_h, nullptr, k2,
                                                           (int)ROWS, DMODEL, DMODEL);
    gemm_tn_kernel<<<gemm_grid_full, 128, 0, stream>>>(hsb, w_tv_h, nullptr, v2t,
                                                       (int)ROWS, DMODEL, DMODEL, SEQ);
    attn_kernel<<<dim3(SEQ / 16, HEADS, BATCH), 160, 0, stream>>>(q2, k2, v2t, mattn, SEQ);
    gemm_kernel<true><<<gemm_grid_full, 128, 0, stream>>>(mattn, w_to_h, b_to, out,
                                                          (int)ROWS, DMODEL, DMODEL);

    // ---- compression branch (reuses q2/k2/v2t/mattn regions) ---------------
    gather_diff_kernel<<<(int)KROWS, 128, 0, stream>>>(hidden, topk, gdiff);
    gemm_kernel<true><<<gemm_grid_half, 128, 0, stream>>>(gdiff, w_primary_h, nullptr,
                                                          comp, (int)KROWS, DMODEL, DMODEL);
    phs_kernel<<<(int)KROWS, 128, 0, stream>>>(comp, b_primary, hidden, topk, phs);
    gemm_kernel<false><<<gemm_grid_half, 128, 0, stream>>>(phs, w_cq_h, b_cq, cq,
                                                           (int)KROWS, DMODEL, DMODEL);
    gemm_kernel<false><<<gemm_grid_full, 128, 0, stream>>>(hsb, w_ck_h, b_ck, ck,
                                                           (int)ROWS, DMODEL, DMODEL);
    gemm_tn_kernel<<<gemm_grid_full, 128, 0, stream>>>(hsb, w_cv_h, b_cv, cvt_t,
                                                       (int)ROWS, DMODEL, DMODEL, SEQ);
    attn_kernel<<<dim3(KTOP / 16, HEADS, BATCH), 160, 0, stream>>>(cq, ck, cvt_t, cattn, KTOP);
    gemm_kernel<true><<<gemm_grid_half, 128, 0, stream>>>(cattn, w_co_h, b_co, comp,
                                                          (int)KROWS, DMODEL, DMODEL);
    cumsum_kernel<<<(KTOP * DMODEL + 255) / 256, 256, 0, stream>>>(comp);
    scatter_add_kernel<<<(int)KROWS, 128, 0, stream>>>(comp, topk, out);
}
